// Encoder_36172214567766
// MI455X (gfx1250) — matrix operand order repaired and checked by reading, not compiled, not run on hardware
//
#include <hip/hip_runtime.h>

// ---------------------------------------------------------------------------
// Types for CDNA5 WMMA (wave32, 16x16x32 f16 -> f32)
// ---------------------------------------------------------------------------
typedef _Float16 hlf;
typedef __attribute__((ext_vector_type(16))) _Float16 v16h;
typedef __attribute__((ext_vector_type(8)))  _Float16 v8h;
typedef __attribute__((ext_vector_type(8)))  float    v8f;
typedef __attribute__((ext_vector_type(4)))  int      v4i;

#define LDA 40  // LDS row stride in halves: 80 bytes, multiple of 16B

#ifndef __has_builtin
#define __has_builtin(x) 0
#endif

// CDNA5 async global->LDS path (ASYNCcnt-tracked), with safe fallback
#if __has_builtin(__builtin_amdgcn_global_load_async_to_lds_b128)
#define HAVE_ASYNC_LDS 1
#else
#define HAVE_ASYNC_LDS 0
#endif

#if __has_builtin(__builtin_amdgcn_s_wait_asynccnt)
#define WAIT_ASYNC(n) __builtin_amdgcn_s_wait_asynccnt(n)
#else
#define WAIT_ASYNC(n) asm volatile("s_wait_asynccnt %0" ::"n"(n) : "memory")
#endif

#define GAS __attribute__((address_space(1)))
#define LAS __attribute__((address_space(3)))
// Builtin signature (from probe diagnostic): (v4i AS1*, v4i AS3*, Ii, Ii)
#define GASP(x) ((GAS v4i*)(x))
#define LASP(x) ((LAS v4i*)(x))

__device__ inline v16h mk16(v8h lo, v8h hi8) {
    v16h r;
#pragma unroll
    for (int i = 0; i < 8; ++i) { r[i] = lo[i]; r[i + 8] = hi8[i]; }
    return r;
}

__device__ inline v8f vzero8() {
    v8f z = {0.f, 0.f, 0.f, 0.f, 0.f, 0.f, 0.f, 0.f};
    return z;
}

// ---------------------------------------------------------------------------
// Weight conversion: f32 W[K][N] -> f16 Wt[N][K]
// ---------------------------------------------------------------------------
__global__ void convt_kernel(const float* __restrict__ W, hlf* __restrict__ Wt,
                             int K, int N) {
    int idx = blockIdx.x * 256 + threadIdx.x;
    if (idx >= K * N) return;
    int k = idx / N, n = idx - k * N;
    Wt[(size_t)n * K + k] = (hlf)W[idx];
}

// ---------------------------------------------------------------------------
// x + positional encoding -> f32 and f16 mirrors
// ---------------------------------------------------------------------------
__global__ void posenc_kernel(const float* __restrict__ x,
                              float* __restrict__ xF, hlf* __restrict__ xH) {
    constexpr int S = 1024, D = 512;
    int idx = blockIdx.x * 256 + threadIdx.x;
    int d = idx & (D - 1);
    int s = (idx / D) & (S - 1);
    int i2 = d & ~1;
    float ang = (float)s * __expf(-(float)i2 * (9.210340371976184f / (float)D));
    float pe = (d & 1) ? __cosf(ang) : __sinf(ang);
    float v = x[idx] + pe;
    xF[idx] = v;
    xH[idx] = (hlf)v;
}

// ---------------------------------------------------------------------------
// Tiled WMMA GEMM, compile-time specialized epilogue.
// out[M,N] = A[M,K](f16) @ Wt[N,K]^T(f16) + bias [+resid] [relu]
// Block: 256 threads = 8 wave32 waves (2 M x 4 N); block tile 128x128;
// wave tile 64x32 = 4x2 WMMA accumulators. Double-buffered async->LDS loads.
// ---------------------------------------------------------------------------
template <bool RES, bool RELU, bool OUTF, bool OUTH>
__global__ __launch_bounds__(256)
void gemm_wmma_kernel(const hlf* __restrict__ A, const hlf* __restrict__ Wt,
                      const float* __restrict__ bias,
                      const float* __restrict__ resid,
                      float* __restrict__ outF, hlf* __restrict__ outH,
                      int N, int K) {
    __shared__ __align__(16) hlf As[2][128 * LDA];
    __shared__ __align__(16) hlf Bs[2][128 * LDA];
    const int tid = threadIdx.x;
    const int lane = tid & 31, w = tid >> 5;
    const int l16 = lane & 15, hi = lane >> 4;
    const int wm = w & 1, wn = w >> 1;
    const int m0 = blockIdx.y * 128, n0 = blockIdx.x * 128;

    // Per-thread tile-load coordinates: 128 rows x 4 16B chunks, 2 per thread
    const int r0 = tid >> 2, c0 = (tid & 3) * 8;
    const int r1 = (tid + 256) >> 2, c1 = c0;

    v8f acc[4][2];
#pragma unroll
    for (int mt = 0; mt < 4; ++mt)
#pragma unroll
        for (int nt = 0; nt < 2; ++nt) acc[mt][nt] = vzero8();

    const int nk = K >> 5;

#if HAVE_ASYNC_LDS
    // ---- CDNA5 async global->LDS double-buffered pipeline ----
    auto issue_tile = [&](int k0, int buf) {
        __builtin_amdgcn_global_load_async_to_lds_b128(
            GASP(&A[(size_t)(m0 + r0) * K + k0 + c0]),
            LASP(&As[buf][r0 * LDA + c0]), 0, 0);
        __builtin_amdgcn_global_load_async_to_lds_b128(
            GASP(&Wt[(size_t)(n0 + r0) * K + k0 + c0]),
            LASP(&Bs[buf][r0 * LDA + c0]), 0, 0);
        __builtin_amdgcn_global_load_async_to_lds_b128(
            GASP(&A[(size_t)(m0 + r1) * K + k0 + c1]),
            LASP(&As[buf][r1 * LDA + c1]), 0, 0);
        __builtin_amdgcn_global_load_async_to_lds_b128(
            GASP(&Wt[(size_t)(n0 + r1) * K + k0 + c1]),
            LASP(&Bs[buf][r1 * LDA + c1]), 0, 0);
    };
    issue_tile(0, 0);
    for (int kt = 0; kt < nk; ++kt) {
        const int buf = kt & 1;
        if (kt + 1 < nk) {
            issue_tile((kt + 1) << 5, buf ^ 1);
            WAIT_ASYNC(4);  // oldest 4 (this tile) complete; next tile in flight
        } else {
            WAIT_ASYNC(0);
        }
        __syncthreads();
        v16h aF[4], bF[2];
#pragma unroll
        for (int mt = 0; mt < 4; ++mt) {
            const hlf* p = &As[buf][(wm * 64 + mt * 16 + l16) * LDA + hi * 8];
            aF[mt] = mk16(*(const v8h*)p, *(const v8h*)(p + 16));
        }
#pragma unroll
        for (int nt = 0; nt < 2; ++nt) {
            const hlf* p = &Bs[buf][(wn * 32 + nt * 16 + l16) * LDA + hi * 8];
            bF[nt] = mk16(*(const v8h*)p, *(const v8h*)(p + 16));
        }
#pragma unroll
        for (int mt = 0; mt < 4; ++mt)
#pragma unroll
            for (int nt = 0; nt < 2; ++nt)
                acc[mt][nt] = __builtin_amdgcn_wmma_f32_16x16x32_f16(
                    false, aF[mt], false, bF[nt], (short)0, acc[mt][nt],
                    false, false);
        __syncthreads();
    }
#else
    // ---- fallback: synchronous load -> LDS ----
    for (int kt = 0; kt < nk; ++kt) {
        const int k0 = kt << 5;
        *(v8h*)&As[0][r0 * LDA + c0] =
            *(const v8h*)&A[(size_t)(m0 + r0) * K + k0 + c0];
        *(v8h*)&Bs[0][r0 * LDA + c0] =
            *(const v8h*)&Wt[(size_t)(n0 + r0) * K + k0 + c0];
        *(v8h*)&As[0][r1 * LDA + c1] =
            *(const v8h*)&A[(size_t)(m0 + r1) * K + k0 + c1];
        *(v8h*)&Bs[0][r1 * LDA + c1] =
            *(const v8h*)&Wt[(size_t)(n0 + r1) * K + k0 + c1];
        __syncthreads();
        v16h aF[4], bF[2];
#pragma unroll
        for (int mt = 0; mt < 4; ++mt) {
            const hlf* p = &As[0][(wm * 64 + mt * 16 + l16) * LDA + hi * 8];
            aF[mt] = mk16(*(const v8h*)p, *(const v8h*)(p + 16));
        }
#pragma unroll
        for (int nt = 0; nt < 2; ++nt) {
            const hlf* p = &Bs[0][(wn * 32 + nt * 16 + l16) * LDA + hi * 8];
            bF[nt] = mk16(*(const v8h*)p, *(const v8h*)(p + 16));
        }
#pragma unroll
        for (int mt = 0; mt < 4; ++mt)
#pragma unroll
            for (int nt = 0; nt < 2; ++nt)
                acc[mt][nt] = __builtin_amdgcn_wmma_f32_16x16x32_f16(
                    false, aF[mt], false, bF[nt], (short)0, acc[mt][nt],
                    false, false);
        __syncthreads();
    }
#endif

    // Epilogue: C/D layout row = r + hi*8, col = l16. All paths static.
#pragma unroll
    for (int mt = 0; mt < 4; ++mt) {
#pragma unroll
        for (int nt = 0; nt < 2; ++nt) {
            const int col = n0 + wn * 32 + nt * 16 + l16;
            const float bv = bias[col];
#pragma unroll
            for (int r = 0; r < 8; ++r) {
                const int row = m0 + wm * 64 + mt * 16 + r + hi * 8;
                const size_t o = (size_t)row * N + col;
                float v = acc[mt][nt][r] + bv;
                if constexpr (RES) v += resid[o];
                if constexpr (RELU) v = fmaxf(v, 0.f);
                if constexpr (OUTF) outF[o] = v;
                if constexpr (OUTH) outH[o] = (hlf)v;
            }
        }
    }
}

// ---------------------------------------------------------------------------
// Flash attention: one block per (b, h, 128-row q block). 8 waves, 16 q rows
// each. Online softmax; Q@K^T and P@V both via v_wmma_f32_16x16x32_f16.
// ---------------------------------------------------------------------------
__global__ __launch_bounds__(256)
void attn_kernel(const hlf* __restrict__ Q, const hlf* __restrict__ Kh,
                 const hlf* __restrict__ V, const int* __restrict__ lens,
                 hlf* __restrict__ O) {
    constexpr int S = 1024, D = 512;
    __shared__ __align__(16) hlf Vt[64 * LDA];      // V block transposed [dh][key]
    __shared__ __align__(16) hlf Pb[8 * 16 * LDA];  // per-wave P staging
    const int tid = threadIdx.x, lane = tid & 31, w = tid >> 5;
    const int l16 = lane & 15, hi = lane >> 4;
    const int b = blockIdx.z, h = blockIdx.y, qb = blockIdx.x;
    const int len = lens[b];
    const int q0 = qb * 128 + w * 16;
    const size_t base = ((size_t)b * S) * D + (size_t)h * 64;

    // Preload Q fragments (A layout), reused across all key blocks
    v16h qf[2];
#pragma unroll
    for (int c = 0; c < 2; ++c) {
        const hlf* p = Q + base + (size_t)(q0 + l16) * D + c * 32 + hi * 8;
        qf[c] = mk16(*(const v8h*)p, *(const v8h*)(p + 16));
    }

    v8f Ov[4];
#pragma unroll
    for (int c2 = 0; c2 < 4; ++c2) Ov[c2] = vzero8();
    float mi[8], li[8];
#pragma unroll
    for (int r = 0; r < 8; ++r) { mi[r] = -1e30f; li[r] = 0.f; }

    for (int kv0 = 0; kv0 < S; kv0 += 32) {
        // Cooperative transpose of V block (32 keys x 64 dh) into LDS
        {
            int key = tid >> 3, ch = tid & 7;
            v8h vv = *(const v8h*)&V[base + (size_t)(kv0 + key) * D + ch * 8];
#pragma unroll
            for (int j = 0; j < 8; ++j) Vt[(ch * 8 + j) * LDA + key] = vv[j];
        }
        __syncthreads();

        // Scores: S(16x32) = Q(16x64) @ K^T, two 16x16 tiles
        v8f sc[2];
        sc[0] = vzero8(); sc[1] = vzero8();
#pragma unroll
        for (int nt = 0; nt < 2; ++nt) {
#pragma unroll
            for (int c = 0; c < 2; ++c) {
                const hlf* p = Kh + base + (size_t)(kv0 + nt * 16 + l16) * D +
                               c * 32 + hi * 8;
                v16h kf = mk16(*(const v8h*)p, *(const v8h*)(p + 16));
                sc[nt] = __builtin_amdgcn_wmma_f32_16x16x32_f16(
                    false, qf[c], false, kf, (short)0, sc[nt], false, false);
            }
        }

        // Online softmax (row stats via 16-lane shfl_xor reductions)
        float pv[2][8];
#pragma unroll
        for (int nt = 0; nt < 2; ++nt) {
            bool ok = (kv0 + nt * 16 + l16) < len;
#pragma unroll
            for (int r = 0; r < 8; ++r)
                pv[nt][r] = ok ? sc[nt][r] * 0.125f : -1e30f;
        }
#pragma unroll
        for (int r = 0; r < 8; ++r) {
            float v = fmaxf(pv[0][r], pv[1][r]);
#pragma unroll
            for (int d = 1; d < 16; d <<= 1) v = fmaxf(v, __shfl_xor(v, d, 32));
            float mn = fmaxf(mi[r], v);
            float a = __expf(mi[r] - mn);
            float p0 = __expf(pv[0][r] - mn);
            float p1 = __expf(pv[1][r] - mn);
            pv[0][r] = p0; pv[1][r] = p1;
            float rs = p0 + p1;
#pragma unroll
            for (int d = 1; d < 16; d <<= 1) rs += __shfl_xor(rs, d, 32);
            li[r] = li[r] * a + rs;
            mi[r] = mn;
#pragma unroll
            for (int c2 = 0; c2 < 4; ++c2) Ov[c2][r] *= a;
        }

        // Stage P through per-wave LDS: C layout -> A layout
        hlf* pb = Pb + w * (16 * LDA);
#pragma unroll
        for (int nt = 0; nt < 2; ++nt)
#pragma unroll
            for (int r = 0; r < 8; ++r)
                pb[(r + hi * 8) * LDA + nt * 16 + l16] = (hlf)pv[nt][r];
        const hlf* pp = pb + l16 * LDA + hi * 8;
        v16h pf = mk16(*(const v8h*)pp, *(const v8h*)(pp + 16));

        // O(16x64) += P(16x32) @ V(32x64)
#pragma unroll
        for (int c2 = 0; c2 < 4; ++c2) {
            const hlf* vp = &Vt[(c2 * 16 + l16) * LDA + hi * 8];
            v16h vf = mk16(*(const v8h*)vp, *(const v8h*)(vp + 16));
            Ov[c2] = __builtin_amdgcn_wmma_f32_16x16x32_f16(
                false, pf, false, vf, (short)0, Ov[c2], false, false);
        }
        __syncthreads();
    }

#pragma unroll
    for (int r = 0; r < 8; ++r) {
        float inv = 1.f / li[r];
#pragma unroll
        for (int c2 = 0; c2 < 4; ++c2)
            O[base + (size_t)(q0 + r + hi * 8) * D + c2 * 16 + l16] =
                (hlf)(Ov[c2][r] * inv);
    }
}

// ---------------------------------------------------------------------------
// LayerNorm (reference semantics: unbiased std, /(std+eps)), one row per block
// ---------------------------------------------------------------------------
__global__ __launch_bounds__(256)
void ln_kernel(const float* __restrict__ in, const float* __restrict__ g,
               const float* __restrict__ be, float* __restrict__ outF,
               hlf* __restrict__ outH) {
    constexpr int D = 512;
    __shared__ float red[256];
    const int row = blockIdx.x, tid = threadIdx.x;
    const size_t o = (size_t)row * D;
    float v0 = in[o + tid], v1 = in[o + 256 + tid];
    red[tid] = v0 + v1;
    __syncthreads();
    for (int s = 128; s > 0; s >>= 1) {
        if (tid < s) red[tid] += red[tid + s];
        __syncthreads();
    }
    float mean = red[0] / (float)D;
    __syncthreads();
    float d0 = v0 - mean, d1 = v1 - mean;
    red[tid] = d0 * d0 + d1 * d1;
    __syncthreads();
    for (int s = 128; s > 0; s >>= 1) {
        if (tid < s) red[tid] += red[tid + s];
        __syncthreads();
    }
    float stdv = sqrtf(red[0] / (float)(D - 1));
    float inv = 1.f / (stdv + 1e-6f);
    float r0 = g[tid] * d0 * inv + be[tid];
    float r1 = g[tid + 256] * d1 * inv + be[tid + 256];
    outF[o + tid] = r0;
    outF[o + 256 + tid] = r1;
    outH[o + tid] = (hlf)r0;
    outH[o + 256 + tid] = (hlf)r1;
}

// ---------------------------------------------------------------------------
// Launch pipeline
// ---------------------------------------------------------------------------
extern "C" void kernel_launch(void* const* d_in, const int* in_sizes, int n_in,
                              void* d_out, int out_size, void* d_ws,
                              size_t ws_size, hipStream_t stream) {
    (void)in_sizes; (void)n_in; (void)out_size; (void)ws_size;
    constexpr int B = 8, S = 1024, D = 512, H = 8, FF = 2048, L = 4;
    constexpr int M = B * S;

    const float* x   = (const float*)d_in[0];
    const int*   lens= (const int*)d_in[1];
    const float* Wq  = (const float*)d_in[2];
    const float* bq  = (const float*)d_in[3];
    const float* Wk  = (const float*)d_in[4];
    const float* bk  = (const float*)d_in[5];
    const float* Wv  = (const float*)d_in[6];
    const float* bv  = (const float*)d_in[7];
    const float* Wo  = (const float*)d_in[8];
    const float* bo  = (const float*)d_in[9];
    const float* W1  = (const float*)d_in[10];
    const float* b1  = (const float*)d_in[11];
    const float* W2  = (const float*)d_in[12];
    const float* b2  = (const float*)d_in[13];
    const float* g1  = (const float*)d_in[14];
    const float* be1 = (const float*)d_in[15];
    const float* g2  = (const float*)d_in[16];
    const float* be2 = (const float*)d_in[17];

    char* p = (char*)d_ws;
    auto take = [&](size_t bytes) -> char* {
        char* r = p;
        p += (bytes + 255) & ~(size_t)255;
        return r;
    };
    float* xF  = (float*)take((size_t)M * D * 4);
    hlf*   xH  = (hlf*)  take((size_t)M * D * 2);
    hlf*   qH  = (hlf*)  take((size_t)M * D * 2);
    hlf*   kH  = (hlf*)  take((size_t)M * D * 2);
    hlf*   vH  = (hlf*)  take((size_t)M * D * 2);
    hlf*   aH  = (hlf*)  take((size_t)M * D * 2);
    float* tF  = (float*)take((size_t)M * D * 4);
    float* yF  = (float*)take((size_t)M * D * 4);
    hlf*   yH  = (hlf*)  take((size_t)M * D * 2);
    hlf*   ffH = (hlf*)  take((size_t)M * FF * 2);
    hlf*   WqT = (hlf*)  take((size_t)L * D * D * 2);
    hlf*   WkT = (hlf*)  take((size_t)L * D * D * 2);
    hlf*   WvT = (hlf*)  take((size_t)L * D * D * 2);
    hlf*   WoT = (hlf*)  take((size_t)L * D * D * 2);
    hlf*   W1T = (hlf*)  take((size_t)L * D * FF * 2);
    hlf*   W2T = (hlf*)  take((size_t)L * FF * D * 2);

    for (int l = 0; l < L; ++l) {
        size_t od = (size_t)l * D * D, of = (size_t)l * D * FF;
        convt_kernel<<<(D * D) / 256, 256, 0, stream>>>(Wq + od, WqT + od, D, D);
        convt_kernel<<<(D * D) / 256, 256, 0, stream>>>(Wk + od, WkT + od, D, D);
        convt_kernel<<<(D * D) / 256, 256, 0, stream>>>(Wv + od, WvT + od, D, D);
        convt_kernel<<<(D * D) / 256, 256, 0, stream>>>(Wo + od, WoT + od, D, D);
        convt_kernel<<<(D * FF) / 256, 256, 0, stream>>>(W1 + of, W1T + of, D, FF);
        convt_kernel<<<(FF * D) / 256, 256, 0, stream>>>(W2 + of, W2T + of, FF, D);
    }

    posenc_kernel<<<(M * D) / 256, 256, 0, stream>>>(x, xF, xH);

    dim3 gD(D / 128, M / 128), gF(FF / 128, M / 128);
    for (int l = 0; l < L; ++l) {
        size_t od = (size_t)l * D * D, of = (size_t)l * D * FF;
        // QKV projections: f16 out only
        gemm_wmma_kernel<false, false, false, true><<<gD, 256, 0, stream>>>(
            xH, WqT + od, bq + l * D, nullptr, nullptr, qH, D, D);
        gemm_wmma_kernel<false, false, false, true><<<gD, 256, 0, stream>>>(
            xH, WkT + od, bk + l * D, nullptr, nullptr, kH, D, D);
        gemm_wmma_kernel<false, false, false, true><<<gD, 256, 0, stream>>>(
            xH, WvT + od, bv + l * D, nullptr, nullptr, vH, D, D);
        attn_kernel<<<dim3(S / 128, H, B), 256, 0, stream>>>(qH, kH, vH, lens, aH);
        // O projection + residual: f32 out
        gemm_wmma_kernel<true, false, true, false><<<gD, 256, 0, stream>>>(
            aH, WoT + od, bo + l * D, xF, tF, nullptr, D, D);
        ln_kernel<<<M, 256, 0, stream>>>(tF, g1 + l * D, be1 + l * D, yF, yH);
        // FFN1 + relu: f16 out
        gemm_wmma_kernel<false, true, false, true><<<gF, 256, 0, stream>>>(
            yH, W1T + of, b1 + l * FF, nullptr, nullptr, ffH, FF, D);
        // FFN2 + residual: f32 out
        gemm_wmma_kernel<true, false, true, false><<<gD, 256, 0, stream>>>(
            ffH, W2T + of, b2 + l * D, yF, tF, nullptr, D, FF);
        float* xo = (l == L - 1) ? (float*)d_out : xF;
        ln_kernel<<<M, 256, 0, stream>>>(tF, g2 + l * D, be2 + l * D, xo, xH);
    }
}
